// MinimalGRU_66202625900890
// MI455X (gfx1250) — matrix operand order repaired and checked by reading, not compiled, not run on hardware
//
#include <hip/hip_runtime.h>
#include <hip/hip_bf16.h>

typedef __attribute__((ext_vector_type(16))) __bf16 v16bf;
typedef __attribute__((ext_vector_type(8)))  __bf16 v8bf;
typedef __attribute__((ext_vector_type(8)))  float  v8f;
typedef int v4i __attribute__((vector_size(16)));  // <4 x i32>, matches builtin proto

#define TSEQ  256
#define BATCH 64
#define HID   1024
#define GATE  2048
#define BK    32
#define LDA   40    // halves stride (80 B, 16B-aligned rows) for As[64][LDA]
#define LDBS  40    // halves stride for phase-1 Bs[32][LDBS]
#define LDBW  1032  // halves stride for persistent Bw[32][LDBW]
#define LDT   33    // float stride for Tile[64][LDT]

#define SM_BW    (32 * LDBW * 2)   // 66048
#define SM_AS    (64 * LDA * 2)    //  5120
#define SM_TILE  (64 * LDT * 4)    //  8448
#define SM_HLOC  (64 * 16 * 4)     //  4096
#define SM_MISC  (6 * 32 * 4)      //   768
#define SM_TOTAL (SM_BW + SM_AS + SM_TILE + SM_HLOC + SM_MISC)

__device__ __forceinline__ __bf16 f2bf(float f) { return (__bf16)f; }

// ---- CDNA5 async global->LDS copy (ASYNCcnt path), sync fallback ----------
#if __has_builtin(__builtin_amdgcn_global_load_async_to_lds_b128)
#define HAVE_ASYNC_LDS 1
#endif

__device__ __forceinline__ void async_cp16(void* lds, const void* g) {
#ifdef HAVE_ASYNC_LDS
  __builtin_amdgcn_global_load_async_to_lds_b128(
      (__attribute__((address_space(1))) v4i*)(size_t)g,
      (__attribute__((address_space(3))) v4i*)lds, 0, 0);
#else
  *(uint4*)lds = *(const uint4*)g;
#endif
}

__device__ __forceinline__ void async_wait0() {
#if __has_builtin(__builtin_amdgcn_s_wait_asynccnt)
  __builtin_amdgcn_s_wait_asynccnt(0);
#elif defined(HAVE_ASYNC_LDS)
  asm volatile("s_wait_asynccnt 0" ::: "memory");
#endif
}

// ---- WMMA fragment loads (ISA 7.12.2 VGPR layouts) ------------------------
// A 16x32 bf16: rows M=lane%16 in both halves; K = 8*half + j&7 + (j>=8)*16
__device__ __forceinline__ v16bf load_a_frag(const __bf16* As, int m0, int lane) {
  v16bf a;
  const int half = lane >> 4;
  const int row  = m0 + (lane & 15);
#pragma unroll
  for (int j = 0; j < 16; ++j) {
    const int k = (j & 7) + (half << 3) + ((j >> 3) << 4);
    a[j] = As[row * LDA + k];
  }
  return a;
}
// B 32x16 bf16 (KxN), stored [n][k]: col N=lane%16; K = 8*half + j&7 + (j>=8)*16
__device__ __forceinline__ v16bf load_b_frag(const __bf16* Bs, int n0, int lane,
                                             int ldb) {
  v16bf b;
  const int half = lane >> 4;
  const int col  = n0 + (lane & 15);
#pragma unroll
  for (int j = 0; j < 16; ++j) {
    const int k = (j & 7) + (half << 3) + ((j >> 3) << 4);
    b[j] = Bs[col * ldb + k];
  }
  return b;
}

// ---- fp32 -> bf16 pack (one-time per layer; amortized over heavy reuse) ---
__global__ __launch_bounds__(256) void cvt_bf16_kernel(const float* __restrict__ in,
                                                       __bf16* __restrict__ out,
                                                       int n8) {
  const int i = blockIdx.x * 256 + threadIdx.x;
  if (i >= n8) return;
  const float4 a = ((const float4*)in)[2 * i + 0];
  const float4 b = ((const float4*)in)[2 * i + 1];
  v8bf v;
  v[0] = f2bf(a.x); v[1] = f2bf(a.y); v[2] = f2bf(a.z); v[3] = f2bf(a.w);
  v[4] = f2bf(b.x); v[5] = f2bf(b.y); v[6] = f2bf(b.z); v[7] = f2bf(b.w);
  ((v8bf*)out)[i] = v;
}

// ---------------------------------------------------------------------------
// Kernel 1: P[t,b,g] = BN_b( x[b,t,:] . W[g,:] + bias[g] ) * bnw[g] + bnb[g]
// grid = (TSEQ, GATE/32), block = 256; M-tile == all 64 batch rows of one t.
// ---------------------------------------------------------------------------
__global__ __launch_bounds__(256, 2) void in_proj_bn_kernel(
    const __bf16* __restrict__ XB, const __bf16* __restrict__ WB,
    const float* __restrict__ bias, const float* __restrict__ bnw,
    const float* __restrict__ bnb, float* __restrict__ P) {
  __shared__ __bf16 As[64 * LDA];
  __shared__ __bf16 Bs[32 * LDBS];
  __shared__ float  Tile[64 * LDT];
  __shared__ float  sMean[32], sRstd[32];

  const int t    = blockIdx.x;
  const int g0   = blockIdx.y << 5;
  const int tid  = threadIdx.x;
  const int lane = tid & 31;
  const int wid  = tid >> 5;
  const int m0   = (wid & 3) << 4;
  const int n0   = (wid >> 2) << 4;
  const int ar   = tid >> 2;        // 0..63 staging row
  const int ac   = (tid & 3) << 3;  // 0,8,16,24 halves (16B chunks)

  v8f acc = {};
  for (int kk = 0; kk < HID; kk += BK) {
    async_cp16(&As[ar * LDA + ac],
               XB + ((size_t)ar * TSEQ + t) * HID + kk + ac);
    if (tid < 128)
      async_cp16(&Bs[(tid >> 2) * LDBS + ac],
                 WB + (size_t)(g0 + (tid >> 2)) * HID + kk + ac);
    async_wait0();
    __syncthreads();
    const v16bf a = load_a_frag(As, m0, lane);
    const v16bf b = load_b_frag(Bs, n0, lane, LDBS);
    acc = __builtin_amdgcn_wmma_f32_16x16x32_bf16(false, a, false, b, (short)0, acc,
                                                  false, false);
    __syncthreads();
  }
  {
    const int half = lane >> 4;
    const int col  = n0 + (lane & 15);
    const float bv = bias[g0 + col];
#pragma unroll
    for (int r = 0; r < 8; ++r)
      Tile[(m0 + r + (half << 3)) * LDT + col] = acc[r] + bv;
  }
  __syncthreads();
  if (tid < 32) {
    float s = 0.f, s2 = 0.f;
#pragma unroll 4
    for (int r = 0; r < 64; ++r) {
      const float v = Tile[r * LDT + tid];
      s += v; s2 += v * v;
    }
    const float m = s * (1.f / 64.f);
    sMean[tid] = m;
    sRstd[tid] = rsqrtf(fmaxf(s2 * (1.f / 64.f) - m * m, 0.f) + 1e-5f);
  }
  __syncthreads();
#pragma unroll
  for (int i = 0; i < 8; ++i) {
    const int idx = tid + (i << 8);
    const int b = idx >> 5, c = idx & 31;
    const float v =
        (Tile[b * LDT + c] - sMean[c]) * sRstd[c] * bnw[g0 + c] + bnb[g0 + c];
    P[((size_t)t * BATCH + b) * GATE + g0 + c] = v;
  }
}

// monotonic grid barrier (counter zeroed via hipMemsetAsync before launch)
__device__ __forceinline__ void grid_sync(unsigned* cnt, unsigned target) {
  __threadfence();
  __syncthreads();
  if (threadIdx.x == 0) {
    __hip_atomic_fetch_add(cnt, 1u, __ATOMIC_ACQ_REL, __HIP_MEMORY_SCOPE_AGENT);
    while (__hip_atomic_load(cnt, __ATOMIC_ACQUIRE, __HIP_MEMORY_SCOPE_AGENT) < target)
      __builtin_amdgcn_s_sleep(2);
  }
  __syncthreads();
  __threadfence();
}

// ---------------------------------------------------------------------------
// Kernel 2: persistent recurrence, 64 blocks. Block owns paired gate columns
// [g0,g0+16) U [H+g0,H+g0+16); whh tile (64KB bf16) persists in LDS; h is a
// bf16 global ping-pong for the GEMM + block-local fp32 hLoc for gating.
// ONE grid barrier per timestep.
// ---------------------------------------------------------------------------
__global__ __launch_bounds__(256, 1) void recurrent_kernel(
    const __bf16* __restrict__ WHB, const float* __restrict__ bhh,
    const float* __restrict__ bnw, const float* __restrict__ bnb,
    const float* __restrict__ P, const float* __restrict__ hx0,
    __bf16* __restrict__ hb0, __bf16* __restrict__ hb1,
    float* __restrict__ OutF, __bf16* __restrict__ OutB,
    unsigned* __restrict__ cnt) {
  extern __shared__ char smem[];
  char* sp = smem;
  __bf16* Bw   = (__bf16*)sp; sp += SM_BW;
  __bf16* As   = (__bf16*)sp; sp += SM_AS;
  float*  Tile = (float*)sp;  sp += SM_TILE;
  float*  hLoc = (float*)sp;  sp += SM_HLOC;
  float*  sMean = (float*)sp;
  float*  sRstd = sMean + 32;
  float*  sBias = sMean + 64;
  float*  sGw   = sMean + 96;
  float*  sGb   = sMean + 128;
  int*    sFeat = (int*)(sMean + 160);

  const int g0   = blockIdx.x << 4;
  const int tid  = threadIdx.x;
  const int lane = tid & 31;
  const int wid  = tid >> 5;
  const int m0   = (wid & 3) << 4;
  const int n0   = (wid >> 2) << 4;
  const int ar   = tid >> 2;
  const int ac   = (tid & 3) << 3;

  if (tid < 32) {
    const int f = (tid < 16) ? (g0 + tid) : (HID + g0 + (tid - 16));
    sFeat[tid] = f;
    sBias[tid] = bhh[f];
    sGw[tid]   = bnw[f];
    sGb[tid]   = bnb[f];
  }
  __syncthreads();

  {  // persist whh tile: 32 feature rows x 1024 k (bf16) -> Bw
    const int n = tid >> 3;
    const int cb = (tid & 7) << 3;
    const size_t gbase = (size_t)sFeat[n] * HID;
#pragma unroll
    for (int it = 0; it < 16; ++it) {
      const int c8 = cb + (it << 6);
      async_cp16(&Bw[n * LDBW + c8], WHB + gbase + c8);
    }
  }
#pragma unroll
  for (int i = 0; i < 4; ++i) {  // fp32 h for our own 16 columns
    const int idx = tid + (i << 8);
    const int b = idx >> 4, c = idx & 15;
    hLoc[b * 16 + c] = hx0[(size_t)b * HID + g0 + c];
  }
  async_wait0();
  __syncthreads();

  unsigned barTarget = 0;
  __bf16* hc = hb0;
  __bf16* hn = hb1;

  for (int t = 0; t < TSEQ; ++t) {
    v8f acc = {};
    for (int kk = 0; kk < HID; kk += BK) {
      async_cp16(&As[ar * LDA + ac], hc + (size_t)ar * HID + kk + ac);
      async_wait0();
      __syncthreads();
      const v16bf a = load_a_frag(As, m0, lane);
      const v16bf b = load_b_frag(Bw + kk, n0, lane, LDBW);
      acc = __builtin_amdgcn_wmma_f32_16x16x32_bf16(false, a, false, b, (short)0, acc,
                                                    false, false);
      __syncthreads();
    }
    {
      const int half = lane >> 4;
      const int col  = n0 + (lane & 15);
      const float bv = sBias[col];
#pragma unroll
      for (int r = 0; r < 8; ++r)
        Tile[(m0 + r + (half << 3)) * LDT + col] = acc[r] + bv;
    }
    __syncthreads();
    if (tid < 32) {
      float s = 0.f, s2 = 0.f;
#pragma unroll 4
      for (int r = 0; r < 64; ++r) {
        const float v = Tile[r * LDT + tid];
        s += v; s2 += v * v;
      }
      const float m = s * (1.f / 64.f);
      sMean[tid] = m;
      sRstd[tid] = rsqrtf(fmaxf(s2 * (1.f / 64.f) - m * m, 0.f) + 1e-5f);
    }
    __syncthreads();
#pragma unroll
    for (int i = 0; i < 8; ++i) {  // gates = BN(h_part) + P (in place in Tile)
      const int idx = tid + (i << 8);
      const int b = idx >> 5, c = idx & 31;
      const float v = (Tile[b * LDT + c] - sMean[c]) * sRstd[c] * sGw[c] + sGb[c] +
                      P[((size_t)t * BATCH + b) * GATE + sFeat[c]];
      Tile[b * LDT + c] = v;
    }
    __syncthreads();
#pragma unroll
    for (int i = 0; i < 4; ++i) {  // gated update, our 16 hidden columns
      const int idx = tid + (i << 8);
      const int b = idx >> 4, c = idx & 15;
      const float gu  = Tile[b * LDT + c];
      const float go  = Tile[b * LDT + 16 + c];
      const float sig = 1.f / (1.f + __expf(-gu));
      const float rl  = fmaxf(go, 0.f);
      const float hv  = hLoc[b * 16 + c];
      const float hnv = sig * hv + (1.f - sig) * rl;
      hLoc[b * 16 + c] = hnv;
      hn[(size_t)b * HID + g0 + c] = f2bf(hnv);
      if (OutF) OutF[((size_t)b * TSEQ + t) * HID + g0 + c] = hnv;
      else      OutB[((size_t)b * TSEQ + t) * HID + g0 + c] = f2bf(hnv);
    }
    if (t + 1 < TSEQ) {  // hint next timestep's P slice toward the caches
      const int pb = tid >> 5, pc = tid & 31;
      __builtin_prefetch(&P[(((size_t)(t + 1)) * BATCH + pb) * GATE + sFeat[pc]], 0, 1);
    }
    barTarget += gridDim.x;
    grid_sync(cnt, barTarget);  // publish h_next to all blocks
    __bf16* tmp = hc; hc = hn; hn = tmp;
  }
}

extern "C" void kernel_launch(void* const* d_in, const int* in_sizes, int n_in,
                              void* d_out, int out_size, void* d_ws, size_t ws_size,
                              hipStream_t stream) {
  const float* x   = (const float*)d_in[0];
  const float* hx  = (const float*)d_in[1];
  const float* wih = (const float*)d_in[2];
  const float* whh = (const float*)d_in[3];
  const float* bih = (const float*)d_in[4];
  const float* bhh = (const float*)d_in[5];
  const float* biw = (const float*)d_in[6];
  const float* bib = (const float*)d_in[7];
  const float* bhw = (const float*)d_in[8];
  const float* bhb = (const float*)d_in[9];

  char* ws = (char*)d_ws;
  float*  P   = (float*)ws;  ws += (size_t)TSEQ * BATCH * GATE * sizeof(float);   // 128MB
  __bf16* XB  = (__bf16*)ws; ws += (size_t)BATCH * TSEQ * HID * sizeof(__bf16);   //  32MB
  __bf16* WIB = (__bf16*)ws; ws += (size_t)GATE * HID * sizeof(__bf16);           //   4MB
  __bf16* WHB = (__bf16*)ws; ws += (size_t)GATE * HID * sizeof(__bf16);           //   4MB
  __bf16* hb0 = (__bf16*)ws; ws += (size_t)BATCH * HID * sizeof(__bf16);
  __bf16* hb1 = (__bf16*)ws; ws += (size_t)BATCH * HID * sizeof(__bf16);
  unsigned* cnt = (unsigned*)ws;

  const int w8 = GATE * HID / 8;          // weight elems / 8
  const int x8 = BATCH * TSEQ * HID / 8;  // activation elems / 8
  const int h8 = BATCH * HID / 8;

  for (int layer = 0; layer < 2; ++layer) {
    const size_t wOff = (size_t)layer * GATE * HID;
    const size_t gOff = (size_t)layer * GATE;
    const float* hx_l = hx + (size_t)layer * BATCH * HID;

    cvt_bf16_kernel<<<w8 / 256, 256, 0, stream>>>(wih + wOff, WIB, w8);
    cvt_bf16_kernel<<<w8 / 256, 256, 0, stream>>>(whh + wOff, WHB, w8);
    if (layer == 0)
      cvt_bf16_kernel<<<x8 / 256, 256, 0, stream>>>(x, XB, x8);
    cvt_bf16_kernel<<<(h8 + 255) / 256, 256, 0, stream>>>(hx_l, hb0, h8);

    in_proj_bn_kernel<<<dim3(TSEQ, GATE / 32), 256, 0, stream>>>(
        XB, WIB, bih + gOff, biw + gOff, bib + gOff, P);

    (void)hipMemsetAsync(cnt, 0, sizeof(unsigned), stream);
    recurrent_kernel<<<dim3(HID / 16), 256, SM_TOTAL, stream>>>(
        WHB, bhh + gOff, bhw + gOff, bhb + gOff, P, hx_l, hb0, hb1,
        (layer == 1) ? (float*)d_out : nullptr,
        (layer == 0) ? XB : nullptr, cnt);
  }
}
